// ImageEncoderTransformer_28329604284442
// MI455X (gfx1250) — matrix operand order repaired and checked by reading, not compiled, not run on hardware
//
#include <hip/hip_runtime.h>
#include <hip/hip_bf16.h>

// ---------------------------------------------------------------------------
// CDNA5 (gfx1250, wave32) implementation of the ImageEncoderTransformer.
// All dense matmuls go through v_wmma_f32_16x16x32_f16 (f32 accum).
// GEMM microkernel: 64x256 block tile, 8 waves each owning 32x64 (8 WMMAs per
// K-step per wave), double-buffered LDS (one barrier per K-step), staging done
// with explicit vector temporaries so global loads clause instead of
// serializing against the LDS stores.
// ---------------------------------------------------------------------------

typedef _Float16 f16;
typedef __attribute__((ext_vector_type(16))) _Float16 v16h;
typedef __attribute__((ext_vector_type(8)))  _Float16 v8h;
typedef __attribute__((ext_vector_type(8)))  float    v8f;
typedef __attribute__((ext_vector_type(4)))  float    v4f;

#define LDSROW 40   // padded LDS row length (f16) for 32-wide K tiles (80B, 16B-aligned)

__device__ __forceinline__ v8f wmma32(v16h a, v16h b, v8f c) {
  // 8 args: (neg_a, A, neg_b, B, c_mod, C, reuse_a, reuse_b)
  return __builtin_amdgcn_wmma_f32_16x16x32_f16(false, a, false, b, (short)0, c,
                                                false, false);
}

// A-matrix fragment (16x32 f16), CDNA5 layout (ISA 7.12.2):
// lane l: M = l&15, g = l>>4 ; VGPR v, elem e: K = (v>>2)*16 + g*8 + (v&3)*2 + e
__device__ __forceinline__ v16h frag_a(const f16* t, int row0, int lane) {
  const f16* p = t + (row0 + (lane & 15)) * LDSROW + ((lane >> 4) << 3);
  v16h f;
#pragma unroll
  for (int v = 0; v < 8; ++v) {
    int kk = ((v >> 2) << 4) + ((v & 3) << 1);
    f[2 * v]     = p[kk];
    f[2 * v + 1] = p[kk + 1];
  }
  return f;
}

// B-matrix fragment (32x16 f16): lane l: N = l&15, g = l>>4 ; K = (i>>3)*16 + g*8 + (i&7)
// Tile is stored N-major in LDS ([N][K]) so each 8-K run is contiguous.
__device__ __forceinline__ v16h frag_b(const f16* t, int row0, int lane) {
  const f16* p = t + (row0 + (lane & 15)) * LDSROW + ((lane >> 4) << 3);
  v16h f;
#pragma unroll
  for (int i = 0; i < 16; ++i) f[i] = p[(i & 7) + ((i >> 3) << 4)];
  return f;
}

// convert 8 f32 (two v4f) -> v8h
__device__ __forceinline__ v8h cvt8(v4f lo, v4f hi) {
  v8h r;
#pragma unroll
  for (int i = 0; i < 4; ++i) { r[i] = (f16)lo[i]; r[i + 4] = (f16)hi[i]; }
  return r;
}

__device__ __forceinline__ int iclip(int v, int lo, int hi) {
  return v < lo ? lo : (v > hi ? hi : v);
}

// ---------------------------------------------------------------------------
// Generic GEMM: C[M,N] = act(A[M,K] @ W[K,N] + bias (+ residual))
// A f32 row-major, W pre-converted f16 transposed ([N][K]).
// Block = 256 threads = 8 waves; tile 64(M) x 256(N), BK=32, double-buffered.
// Requires M%64==0, N%256==0, K%32==0 (true for every call site).
// ---------------------------------------------------------------------------
__global__ __launch_bounds__(256)
void gemm_wt_kernel(const float* __restrict__ A, const f16* __restrict__ Wt,
                    const float* __restrict__ bias, const float* __restrict__ resid,
                    float* __restrict__ C, int M, int N, int Kd, int relu) {
  __shared__ f16 As[2][64 * LDSROW];
  __shared__ f16 Bs[2][256 * LDSROW];
  const int tid = threadIdx.x, lane = tid & 31, w = tid >> 5;
  const int wm = (w & 1) * 32;   // 2 waves tile M: 32 rows each
  const int wn = (w >> 1) * 64;  // 4 waves tile N: 64 cols each
  const int m0 = blockIdx.x * 64;
  const int n0 = blockIdx.y * 256;
  v8f acc[2][4];
#pragma unroll
  for (int i = 0; i < 2; ++i)
#pragma unroll
    for (int j = 0; j < 4; ++j) acc[i][j] = (v8f){};

  const int ar = tid >> 2, ac = (tid & 3) * 8;  // A staging: 8 f32 / thread
  const int nk = Kd / 32;

  auto stage = [&](int buf, int kb) {
    const int k0 = kb * 32;
    // --- A: 8 f32 -> f16 per thread, explicit vector temps ---
    const v4f* ap = (const v4f*)(A + (size_t)(m0 + ar) * Kd + k0 + ac);
    __builtin_prefetch((const void*)(ap + 8), 0, 1);  // global_prefetch_b8
    v4f a_lo = ap[0];
    v4f a_hi = ap[1];
    // --- B: one full Wt row (64B contiguous) per thread, 4 distinct temps ---
    const v8h* bp = (const v8h*)(Wt + (size_t)(n0 + tid) * Kd + k0);
    v8h b0 = bp[0], b1 = bp[1], b2 = bp[2], b3 = bp[3];
    // stores after all loads are issued (lets the loads clause)
    *(v8h*)(As[buf] + ar * LDSROW + ac) = cvt8(a_lo, a_hi);
    v8h* db = (v8h*)(Bs[buf] + tid * LDSROW);
    db[0] = b0; db[1] = b1; db[2] = b2; db[3] = b3;
  };

  stage(0, 0);
  for (int kb = 0; kb < nk; ++kb) {
    __syncthreads();  // staging of current buffer complete (incl. s_wait_dscnt)
    if (kb + 1 < nk) stage((kb + 1) & 1, kb + 1);
    const f16* as = As[kb & 1];
    const f16* bs = Bs[kb & 1];
    v16h a0 = frag_a(as, wm, lane);
    v16h a1 = frag_a(as, wm + 16, lane);
#pragma unroll
    for (int j = 0; j < 4; ++j) {
      v16h bf = frag_b(bs, wn + 16 * j, lane);
      acc[0][j] = wmma32(a0, bf, acc[0][j]);
      acc[1][j] = wmma32(a1, bf, acc[1][j]);
    }
  }
  // C layout: lane l: N = l&15, M = r + 8*(l>>4)
  const int nlane = lane & 15, mg = (lane >> 4) * 8;
#pragma unroll
  for (int i = 0; i < 2; ++i) {
    const int gm = m0 + wm + i * 16 + mg;
#pragma unroll
    for (int j = 0; j < 4; ++j) {
      const int gn = n0 + wn + j * 16 + nlane;
      float bv = bias ? bias[gn] : 0.0f;
#pragma unroll
      for (int r = 0; r < 8; ++r) {
        float v = acc[i][j][r] + bv;
        if (resid) v += resid[(size_t)(gm + r) * N + gn];
        if (relu) v = fmaxf(v, 0.0f);
        C[(size_t)(gm + r) * N + gn] = v;
      }
    }
  }
}

// ---------------------------------------------------------------------------
// Attention scores:  Sc[b,h,q,k] = ( Q.K^T + preloaded qS ) * scale
// Sc must be pre-filled with the q.S structure term.
// grid = (L/64, L/64, B*nh); K-loop is hd=64 (2 steps) -> single buffered.
// ---------------------------------------------------------------------------
__global__ __launch_bounds__(256)
void attn_qk_kernel(const float* __restrict__ Q, const float* __restrict__ Km,
                    float* __restrict__ Sc, int L, int D, int nh, int hd,
                    float scale) {
  __shared__ f16 As[64 * LDSROW];
  __shared__ f16 Bs[64 * LDSROW];
  const int tid = threadIdx.x, lane = tid & 31, w = tid >> 5;
  const int wm = (w & 3) * 16, wn = (w >> 2) * 32;
  const int m0 = blockIdx.x * 64, n0 = blockIdx.y * 64;
  const int bh = blockIdx.z, b = bh / nh, h = bh % nh;
  const float* qb = Q + (size_t)b * L * D + (size_t)h * hd;
  const float* kb = Km + (size_t)b * L * D + (size_t)h * hd;
  float* sb = Sc + (size_t)bh * L * L;
  const int nlane = lane & 15, mg = (lane >> 4) * 8;
  v8f c0, c1;
#pragma unroll
  for (int r = 0; r < 8; ++r) {
    c0[r] = sb[(size_t)(m0 + wm + mg + r) * L + (n0 + wn + nlane)];
    c1[r] = sb[(size_t)(m0 + wm + mg + r) * L + (n0 + wn + 16 + nlane)];
  }
  const int sr = tid >> 2, sc = (tid & 3) * 8;
  for (int k0 = 0; k0 < hd; k0 += 32) {
    const v4f* ap = (const v4f*)(qb + (size_t)(m0 + sr) * D + k0 + sc);
    const v4f* bp = (const v4f*)(kb + (size_t)(n0 + sr) * D + k0 + sc);
    v4f a_lo = ap[0], a_hi = ap[1];
    v4f b_lo = bp[0], b_hi = bp[1];
    *(v8h*)(As + sr * LDSROW + sc) = cvt8(a_lo, a_hi);
    *(v8h*)(Bs + sr * LDSROW + sc) = cvt8(b_lo, b_hi);
    __syncthreads();
    v16h af = frag_a(As, wm, lane);
    v16h b0 = frag_b(Bs, wn, lane);
    v16h b1 = frag_b(Bs, wn + 16, lane);
    c0 = wmma32(af, b0, c0);
    c1 = wmma32(af, b1, c1);
    __syncthreads();
  }
#pragma unroll
  for (int r = 0; r < 8; ++r) {
    sb[(size_t)(m0 + wm + mg + r) * L + (n0 + wn + nlane)]      = c0[r] * scale;
    sb[(size_t)(m0 + wm + mg + r) * L + (n0 + wn + 16 + nlane)] = c1[r] * scale;
  }
}

// ---------------------------------------------------------------------------
// Attention output: O[b,q,h,:] = P.V + preloaded attn.S term (in O).
// N dim = hd = 64. grid = (L/64, 1, B*nh)
// ---------------------------------------------------------------------------
__global__ __launch_bounds__(256)
void attn_av_kernel(const float* __restrict__ P, const float* __restrict__ V,
                    float* __restrict__ O, int L, int D, int nh, int hd) {
  __shared__ f16 As[64 * LDSROW];
  __shared__ f16 Bs[64 * LDSROW];
  const int tid = threadIdx.x, lane = tid & 31, w = tid >> 5;
  const int wm = (w & 3) * 16, wn = (w >> 2) * 32;
  const int m0 = blockIdx.x * 64;
  const int bh = blockIdx.z, b = bh / nh, h = bh % nh;
  const float* pb = P + (size_t)bh * L * L;
  const float* vb = V + (size_t)b * L * D + (size_t)h * hd;
  float* ob = O + (size_t)b * L * D + (size_t)h * hd;
  const int nlane = lane & 15, mg = (lane >> 4) * 8;
  v8f c0, c1;
#pragma unroll
  for (int r = 0; r < 8; ++r) {
    c0[r] = ob[(size_t)(m0 + wm + mg + r) * D + (wn + nlane)];
    c1[r] = ob[(size_t)(m0 + wm + mg + r) * D + (wn + 16 + nlane)];
  }
  for (int k0 = 0; k0 < L; k0 += 32) {
    {  // stage attn rows [64 q][32 k] -> f16 (vector temps)
      int sr = tid >> 2, sc = (tid & 3) * 8;
      const v4f* ap = (const v4f*)(pb + (size_t)(m0 + sr) * L + k0 + sc);
      v4f a_lo = ap[0], a_hi = ap[1];
      *(v8h*)(As + sr * LDSROW + sc) = cvt8(a_lo, a_hi);
    }
    {  // stage V transposed: Bs[n][kk] = V[k0+kk][n]
      int kk = tid >> 3, nb = (tid & 7) * 8;
      const v4f* vp = (const v4f*)(vb + (size_t)(k0 + kk) * D + nb);
      v4f v_lo = vp[0], v_hi = vp[1];
#pragma unroll
      for (int j = 0; j < 4; ++j) {
        Bs[(nb + j) * LDSROW + kk]     = (f16)v_lo[j];
        Bs[(nb + 4 + j) * LDSROW + kk] = (f16)v_hi[j];
      }
    }
    __syncthreads();
    v16h af = frag_a(As, wm, lane);
    v16h b0 = frag_b(Bs, wn, lane);
    v16h b1 = frag_b(Bs, wn + 16, lane);
    c0 = wmma32(af, b0, c0);
    c1 = wmma32(af, b1, c1);
    __syncthreads();
  }
#pragma unroll
  for (int r = 0; r < 8; ++r) {
    ob[(size_t)(m0 + wm + mg + r) * D + (wn + nlane)]      = c0[r];
    ob[(size_t)(m0 + wm + mg + r) * D + (wn + 16 + nlane)] = c1[r];
  }
}

// ---------------------------------------------------------------------------
// VALU structure-term kernels
// ---------------------------------------------------------------------------
// Sc[b,h,q,k] = sum_d Q[b,q,h,d] * S[q,k,d]
__global__ void qs_kernel(const float* __restrict__ Q, const float* __restrict__ S,
                          float* __restrict__ Sc, int Bp, int L, int D, int nh,
                          int hd) {
  long long idx = (long long)blockIdx.x * 256 + threadIdx.x;
  long long total = (long long)Bp * nh * L * L;
  if (idx >= total) return;
  int k = (int)(idx % L);
  long long r = idx / L;
  int q = (int)(r % L);
  int bh = (int)(r / L);
  int b = bh / nh, h = bh % nh;
  const float* qp = Q + ((size_t)(b * L + q)) * D + h * hd;
  const float* sp = S + ((size_t)q * L + k) * hd;
  float acc = 0.f;
  for (int d = 0; d < hd; ++d) acc += qp[d] * sp[d];
  Sc[idx] = acc;
}

// O[b,q,h,dd] = sum_k P[b,h,q,k] * S[q,k,dd]
__global__ void attns_kernel(const float* __restrict__ P, const float* __restrict__ S,
                             float* __restrict__ O, int Bp, int L, int D, int nh,
                             int hd) {
  long long idx = (long long)blockIdx.x * 256 + threadIdx.x;
  long long total = (long long)Bp * L * nh * hd;
  if (idx >= total) return;
  int dd = (int)(idx % hd);
  long long r = idx / hd;
  int h = (int)(r % nh);
  r /= nh;
  int q = (int)(r % L);
  int b = (int)(r / L);
  const float* ap = P + ((size_t)(b * nh + h) * L + q) * L;
  const float* sp = S + (size_t)q * L * hd + dd;
  float acc = 0.f;
  for (int k = 0; k < L; ++k) acc += ap[k] * sp[(size_t)k * hd];
  O[((size_t)(b * L + q)) * D + h * hd + dd] = acc;
}

// row softmax over 256 entries; one wave32 per row, 8 values per lane
__global__ __launch_bounds__(256)
void softmax256_kernel(float* __restrict__ S, long long nrows) {
  long long row = (long long)blockIdx.x * 8 + (threadIdx.x >> 5);
  if (row >= nrows) return;
  int lane = threadIdx.x & 31;
  float* p = S + row * 256;
  float v[8];
  float mx = -3.0e38f;
#pragma unroll
  for (int j = 0; j < 8; ++j) { v[j] = p[lane + 32 * j]; mx = fmaxf(mx, v[j]); }
#pragma unroll
  for (int off = 16; off; off >>= 1) mx = fmaxf(mx, __shfl_xor(mx, off, 32));
  float sum = 0.f;
#pragma unroll
  for (int j = 0; j < 8; ++j) { v[j] = __expf(v[j] - mx); sum += v[j]; }
#pragma unroll
  for (int off = 16; off; off >>= 1) sum += __shfl_xor(sum, off, 32);
  float inv = 1.0f / sum;
#pragma unroll
  for (int j = 0; j < 8; ++j) p[lane + 32 * j] = v[j] * inv;
}

// LayerNorm: block per row
__global__ __launch_bounds__(256)
void layernorm_kernel(const float* __restrict__ X, float* __restrict__ Y,
                      const float* __restrict__ g, const float* __restrict__ b,
                      int D) {
  int row = blockIdx.x;
  const float* x = X + (size_t)row * D;
  float* y = Y + (size_t)row * D;
  int tid = threadIdx.x;
  float s = 0.f, s2 = 0.f;
  for (int i = tid; i < D; i += 256) { float v = x[i]; s += v; s2 += v * v; }
#pragma unroll
  for (int off = 16; off; off >>= 1) {
    s += __shfl_xor(s, off, 32);
    s2 += __shfl_xor(s2, off, 32);
  }
  __shared__ float rs[8], rq[8];
  __shared__ float mean_s, rstd_s;
  int w = tid >> 5;
  if ((tid & 31) == 0) { rs[w] = s; rq[w] = s2; }
  __syncthreads();
  if (tid == 0) {
    float ts = 0.f, tq = 0.f;
    for (int i = 0; i < 8; ++i) { ts += rs[i]; tq += rq[i]; }
    float mean = ts / D;
    float var = tq / D - mean * mean;
    mean_s = mean;
    rstd_s = rsqrtf(var + 1e-5f);
  }
  __syncthreads();
  float mean = mean_s, rstd = rstd_s;
  for (int i = tid; i < D; i += 256) y[i] = (x[i] - mean) * rstd * g[i] + b[i];
}

// Patchify + pixel projection: h[(p*256+t)*256+dd]
__global__ void patch_proj_kernel(const float* __restrict__ x,
                                  const float* __restrict__ W,
                                  const float* __restrict__ bias,
                                  float* __restrict__ h) {
  long long idx = (long long)blockIdx.x * 256 + threadIdx.x;
  if (idx >= (long long)256 * 256 * 256) return;
  int dd = (int)(idx & 255);
  long long r = idx >> 8;
  int t = (int)(r & 255);
  int p = (int)(r >> 8);
  int px = p & 15, py = p >> 4, tx = t & 15, ty = t >> 4;
  int iy = py * 16 + ty, ix = px * 16 + tx;
  const float* xp = x + iy * 256 + ix;  // channel stride 65536
  float acc = bias[dd];
#pragma unroll
  for (int c = 0; c < 3; ++c) acc += xp[c * 65536] * W[c * 256 + dd];
  h[idx] = acc;
}

// relative 2D position structure matrix S[i,j,dd]
__global__ void pos2d_kernel(const float* __restrict__ xe,
                             const float* __restrict__ ye, float* __restrict__ S,
                             int sz, int maxs, int d) {
  long long idx = (long long)blockIdx.x * 256 + threadIdx.x;
  long long total = (long long)sz * sz * sz * sz * d;
  if (idx >= total) return;
  int dd = (int)(idx % d);
  long long r = idx / d;
  int n2 = sz * sz;
  int j = (int)(r % n2);
  int i = (int)(r / n2);
  int xi = i % sz, yi = i / sz, xj = j % sz, yj = j / sz;
  int xd = iclip(xj - xi, -maxs, maxs) + maxs;
  int yd = iclip(yj - yi, -maxs, maxs) + maxs;
  S[idx] = xe[(size_t)xd * d + dd] + ye[(size_t)yd * d + dd];
}

// S1d[k,dd] = pe1d_x[0,dd] + pe1d_y[k,dd]
__global__ void s1d_kernel(const float* __restrict__ xe,
                           const float* __restrict__ ye, float* __restrict__ S,
                           int Lk, int d) {
  int idx = blockIdx.x * 256 + threadIdx.x;
  if (idx >= Lk * d) return;
  int dd = idx % d, k = idx / d;
  S[idx] = xe[dd] + ye[(size_t)k * d + dd];
}

// W[K,N] f32 -> Wt[N,K] f16
__global__ void cvt_w_kernel(const float* __restrict__ W, f16* __restrict__ Wt,
                             int Kd, int N) {
  long long idx = (long long)blockIdx.x * 256 + threadIdx.x;
  if (idx >= (long long)Kd * N) return;
  int k = (int)(idx % Kd);
  long long n = idx / Kd;
  Wt[idx] = (f16)W[(size_t)k * N + n];
}

// Y[g,dd] = mean_t X[g*T+t, dd]
__global__ void mean_rows_kernel(const float* __restrict__ X, float* __restrict__ Y,
                                 int groups, int T, int D) {
  int idx = blockIdx.x * 256 + threadIdx.x;
  if (idx >= groups * D) return;
  int dd = idx % D, g = idx / D;
  const float* p = X + (size_t)g * T * D + dd;
  float acc = 0.f;
  for (int t = 0; t < T; ++t) acc += p[(size_t)t * D];
  Y[idx] = acc / (float)T;
}

// cross-attn scores (query length 1): sc[(b*nh+h)*Lk + k]
__global__ void cross_scores_kernel(const float* __restrict__ Q,
                                    const float* __restrict__ Kc,
                                    const float* __restrict__ S1d,
                                    float* __restrict__ out, int Bp, int Lk, int D,
                                    int nh, int hd, float scale) {
  int idx = blockIdx.x * 256 + threadIdx.x;
  if (idx >= Bp * nh * Lk) return;
  int k = idx % Lk, bh = idx / Lk;
  int h = bh % nh, b = bh / nh;
  const float* qp = Q + (size_t)b * D + h * hd;
  const float* kp = Kc + ((size_t)b * Lk + k) * D + h * hd;
  const float* sp = S1d + (size_t)k * hd;
  float acc = 0.f;
  for (int d = 0; d < hd; ++d) acc += qp[d] * (kp[d] + sp[d]);
  out[idx] = acc * scale;
}

// cross-attn output: o[b, h*hd+dd]
__global__ void cross_av_kernel(const float* __restrict__ P,
                                const float* __restrict__ V,
                                const float* __restrict__ S1d,
                                float* __restrict__ out, int Bp, int Lk, int D,
                                int nh, int hd) {
  int idx = blockIdx.x * 256 + threadIdx.x;
  if (idx >= Bp * nh * hd) return;
  int dd = idx % hd, bh = idx / hd;
  int h = bh % nh, b = bh / nh;
  const float* ap = P + (size_t)bh * Lk;
  const float* vp = V + (size_t)b * Lk * D + h * hd + dd;
  const float* sp = S1d + dd;
  float acc = 0.f;
  for (int k = 0; k < Lk; ++k)
    acc += ap[k] * (vp[(size_t)k * D] + sp[(size_t)k * hd]);
  out[(size_t)b * D + h * hd + dd] = acc;
}

// ---------------------------------------------------------------------------
// Host orchestration
// ---------------------------------------------------------------------------
struct LayerW {
  f16 *wq, *wk, *wv, *wo, *w1, *w2;
  const float *bq, *bk, *bv, *bo, *b1, *b2, *ln1g, *ln1b, *ln2g, *ln2b;
};

extern "C" void kernel_launch(void* const* d_in, const int* in_sizes, int n_in,
                              void* d_out, int out_size, void* d_ws, size_t ws_size,
                              hipStream_t stream) {
  (void)n_in; (void)out_size; (void)ws_size;
  const dim3 blk(256);
  auto get = [&](int i) { return (const float*)d_in[i]; };

  // ---- input index mapping (insertion order vs sorted pytree order) ----
  // enum order: WQ,BQ,WK,BK,WV,BV,WO,BO,LN1G,LN1B,W1,B1,W2,B2,LN2G,LN2B
  static const int LPA[16] = {0, 1, 2, 3, 4, 5, 6, 7, 8, 9, 10, 11, 12, 13, 14, 15};
  static const int LPB[16] = {4, 10, 2, 8, 5, 11, 3, 9, 13, 12, 0, 6, 1, 7, 15, 14};
  bool orderA = (in_sizes[0] == 196608);  // x first => insertion order
  const int* LP = orderA ? LPA : LPB;
  int iX, iPixW, iPixB, iPe1x, iPe1y, iOut1W, iOut1B, iPe1dx, iPe1dy, iPe2x, iPe2y;
  int bEnc1[2], bCross, bEnc2[4];
  if (orderA) {
    iX = 0; iPixW = 1; iPixB = 2; iPe1x = 3; iPe1y = 4;
    bEnc1[0] = 5; bEnc1[1] = 21;
    iOut1W = 37; iOut1B = 38; iPe1dx = 39; iPe1dy = 40;
    bCross = 41; iPe2x = 57; iPe2y = 58;
    bEnc2[0] = 59; bEnc2[1] = 75; bEnc2[2] = 91; bEnc2[3] = 107;
  } else {
    bCross = 0; bEnc1[0] = 16; bEnc1[1] = 32;
    bEnc2[0] = 48; bEnc2[1] = 64; bEnc2[2] = 80; bEnc2[3] = 96;
    iOut1W = 112; iOut1B = 113; iPe1x = 114; iPe1y = 115;
    iPe1dx = 116; iPe1dy = 117; iPe2x = 118; iPe2y = 119;
    iPixW = 120; iPixB = 121; iX = 122;
  }

  // ---- workspace carve-up ----
  char* base = (char*)d_ws;
  size_t o = 0;
  auto take = [&](size_t bytes) -> char* {
    char* p = base + o;
    o += (bytes + 255) & ~(size_t)255;
    return p;
  };
  f16* wtpool  = (f16*)take(19000000ull * 2);            // f16 weight pool (~36MB)
  float* S1    = (float*)take(4194304ull * 4);           // [256,256,64]
  float* S2    = (float*)take(4194304ull * 4);           // [256,256,64]
  float* S1d   = (float*)take(16384ull * 4);             // [256,64]
  float* h512  = (float*)take(65536ull * 512 * 4);       // post-out1 activations
  float* big   = (float*)take(268435456ull);             // scores / FFN-mid (shared)
  float* h1    = (float*)take(65536ull * 256 * 4);       // enc1 activations
  float* tmp1  = (float*)take(65536ull * 256 * 4);
  float* qb    = (float*)take(65536ull * 512 * 4);       // enc1 Q / cross K (134MB)
  float* kb    = (float*)take(65536ull * 512 * 4);       // enc1 K / cross V (134MB)
  float* vb    = (float*)take(65536ull * 256 * 4);
  float* ao    = (float*)take(65536ull * 256 * 4);
  float* hm    = (float*)take(256ull * 512 * 4);
  float* q2    = (float*)take(256ull * 512 * 4);
  float* ao2   = (float*)take(256ull * 512 * 4);
  float* tmp2  = (float*)take(256ull * 512 * 4);
  float* sc2   = (float*)take(2048ull * 256 * 4);
  float* h2    = (float*)take(256ull * 512 * 4);
  float* q3    = (float*)take(256ull * 512 * 4);
  float* k3    = (float*)take(256ull * 512 * 4);
  float* v3    = (float*)take(256ull * 512 * 4);
  float* ao3   = (float*)take(256ull * 512 * 4);
  float* tmp3  = (float*)take(256ull * 512 * 4);

  // ---- weight conversion (f32 [K,N] -> f16 [N,K]) ----
  f16* wcur = wtpool;
  auto cvt = [&](const float* W, int Kd, int N) -> f16* {
    f16* dst = wcur;
    long long total = (long long)Kd * N;
    wcur += total;
    cvt_w_kernel<<<(unsigned)((total + 255) / 256), blk, 0, stream>>>(W, dst, Kd, N);
    return dst;
  };
  auto mkLayer = [&](int bidx, int D) -> LayerW {
    LayerW wl;
    int Dff = 4 * D;
    wl.wq = cvt(get(bidx + LP[0]), D, D);   wl.bq = get(bidx + LP[1]);
    wl.wk = cvt(get(bidx + LP[2]), D, D);   wl.bk = get(bidx + LP[3]);
    wl.wv = cvt(get(bidx + LP[4]), D, D);   wl.bv = get(bidx + LP[5]);
    wl.wo = cvt(get(bidx + LP[6]), D, D);   wl.bo = get(bidx + LP[7]);
    wl.ln1g = get(bidx + LP[8]);            wl.ln1b = get(bidx + LP[9]);
    wl.w1 = cvt(get(bidx + LP[10]), D, Dff); wl.b1 = get(bidx + LP[11]);
    wl.w2 = cvt(get(bidx + LP[12]), Dff, D); wl.b2 = get(bidx + LP[13]);
    wl.ln2g = get(bidx + LP[14]);           wl.ln2b = get(bidx + LP[15]);
    return wl;
  };
  LayerW enc1[2] = {mkLayer(bEnc1[0], 256), mkLayer(bEnc1[1], 256)};
  f16* wtOut1 = cvt(get(iOut1W), 256, 512);
  LayerW cross = mkLayer(bCross, 512);
  LayerW enc2[4] = {mkLayer(bEnc2[0], 512), mkLayer(bEnc2[1], 512),
                    mkLayer(bEnc2[2], 512), mkLayer(bEnc2[3], 512)};

  // ---- structure matrices ----
  pos2d_kernel<<<(4194304 + 255) / 256, blk, 0, stream>>>(get(iPe1x), get(iPe1y), S1, 16, 16, 64);
  pos2d_kernel<<<(4194304 + 255) / 256, blk, 0, stream>>>(get(iPe2x), get(iPe2y), S2, 16, 256, 64);
  s1d_kernel<<<(16384 + 255) / 256, blk, 0, stream>>>(get(iPe1dx), get(iPe1dy), S1d, 256, 64);

  // ---- patchify + pixel projection -> h1 [65536, 256] ----
  patch_proj_kernel<<<(16777216 + 255) / 256, blk, 0, stream>>>(get(iX), get(iPixW), get(iPixB), h1);

  // ---- generic encoder layer (self-attn + FFN), in-place on xbuf ----
  auto run_layer = [&](float* xbuf, int Bp, int L, int D, int nh, const float* S,
                       const LayerW& Lw, float* q, float* k, float* v,
                       float* scores, float* aobuf, float* mid, float* tmp) {
    int rows = Bp * L, hd = D / nh, Dff = 4 * D;
    dim3 gP(rows / 64, D / 256);
    gemm_wt_kernel<<<gP, blk, 0, stream>>>(xbuf, Lw.wq, Lw.bq, nullptr, q, rows, D, D, 0);
    gemm_wt_kernel<<<gP, blk, 0, stream>>>(xbuf, Lw.wk, Lw.bk, nullptr, k, rows, D, D, 0);
    gemm_wt_kernel<<<gP, blk, 0, stream>>>(xbuf, Lw.wv, Lw.bv, nullptr, v, rows, D, D, 0);
    long long tqs = (long long)Bp * nh * L * L;
    qs_kernel<<<(unsigned)((tqs + 255) / 256), blk, 0, stream>>>(q, S, scores, Bp, L, D, nh, hd);
    dim3 gQK(L / 64, L / 64, Bp * nh);
    attn_qk_kernel<<<gQK, blk, 0, stream>>>(q, k, scores, L, D, nh, hd, 0.125f);
    long long nrows = (long long)Bp * nh * L;
    softmax256_kernel<<<(unsigned)((nrows + 7) / 8), blk, 0, stream>>>(scores, nrows);
    long long tas = (long long)rows * D;
    attns_kernel<<<(unsigned)((tas + 255) / 256), blk, 0, stream>>>(scores, S, aobuf, Bp, L, D, nh, hd);
    dim3 gAV(L / 64, 1, Bp * nh);
    attn_av_kernel<<<gAV, blk, 0, stream>>>(scores, v, aobuf, L, D, nh, hd);
    gemm_wt_kernel<<<gP, blk, 0, stream>>>(aobuf, Lw.wo, Lw.bo, xbuf, tmp, rows, D, D, 0);
    layernorm_kernel<<<rows, blk, 0, stream>>>(tmp, xbuf, Lw.ln1g, Lw.ln1b, D);
    dim3 gF1(rows / 64, Dff / 256);
    gemm_wt_kernel<<<gF1, blk, 0, stream>>>(xbuf, Lw.w1, Lw.b1, nullptr, mid, rows, Dff, D, 1);
    dim3 gF2(rows / 64, D / 256);
    gemm_wt_kernel<<<gF2, blk, 0, stream>>>(mid, Lw.w2, Lw.b2, xbuf, tmp, rows, D, Dff, 0);
    layernorm_kernel<<<rows, blk, 0, stream>>>(tmp, xbuf, Lw.ln2g, Lw.ln2b, D);
  };

  // ---- enc1: 256 patch-transformers, L=256, D=256, nh=4 ----
  for (int l = 0; l < 2; ++l)
    run_layer(h1, 256, 256, 256, 4, S1, enc1[l], qb, kb, vb, big, ao, big, tmp1);

  // ---- out1: [65536,256] @ [256,512] -> h512 ----
  {
    dim3 g(65536 / 64, 512 / 256);
    gemm_wt_kernel<<<g, blk, 0, stream>>>(h1, wtOut1, get(iOut1B), nullptr, h512, 65536, 512, 256, 0);
  }

  // ---- cross layer: q = mean over tokens, kv = h512 ----
  mean_rows_kernel<<<(256 * 512 + 255) / 256, blk, 0, stream>>>(h512, hm, 256, 256, 512);
  {
    dim3 gq(256 / 64, 512 / 256), gkv(65536 / 64, 512 / 256);
    gemm_wt_kernel<<<gq, blk, 0, stream>>>(hm, cross.wq, cross.bq, nullptr, q2, 256, 512, 512, 0);
    gemm_wt_kernel<<<gkv, blk, 0, stream>>>(h512, cross.wk, cross.bk, nullptr, qb, 65536, 512, 512, 0);
    gemm_wt_kernel<<<gkv, blk, 0, stream>>>(h512, cross.wv, cross.bv, nullptr, kb, 65536, 512, 512, 0);
    cross_scores_kernel<<<(256 * 8 * 256 + 255) / 256, blk, 0, stream>>>(
        q2, qb, S1d, sc2, 256, 256, 512, 8, 64, 0.125f);
    softmax256_kernel<<<(2048 + 7) / 8, blk, 0, stream>>>(sc2, 2048);
    cross_av_kernel<<<(256 * 8 * 64 + 255) / 256, blk, 0, stream>>>(
        sc2, kb, S1d, ao2, 256, 256, 512, 8, 64);
    gemm_wt_kernel<<<gq, blk, 0, stream>>>(ao2, cross.wo, cross.bo, hm, tmp2, 256, 512, 512, 0);
    layernorm_kernel<<<256, blk, 0, stream>>>(tmp2, h2, cross.ln1g, cross.ln1b, 512);
    dim3 gf1(256 / 64, 2048 / 256), gf2(256 / 64, 512 / 256);
    gemm_wt_kernel<<<gf1, blk, 0, stream>>>(h2, cross.w1, cross.b1, nullptr, big, 256, 2048, 512, 1);
    gemm_wt_kernel<<<gf2, blk, 0, stream>>>(big, cross.w2, cross.b2, h2, tmp2, 256, 512, 2048, 0);
    layernorm_kernel<<<256, blk, 0, stream>>>(tmp2, h2, cross.ln2g, cross.ln2b, 512);
  }

  // ---- enc2: one transformer, L=256, D=512, nh=8 ----
  for (int l = 0; l < 4; ++l)
    run_layer(h2, 1, 256, 512, 8, S2, enc2[l], q3, k3, v3, big, ao3, big, tmp3);

  // ---- output: h2 [256,512] == [1,16,16,512] ----
  hipMemcpyAsync(d_out, h2, 256ull * 512 * sizeof(float), hipMemcpyDeviceToDevice, stream);
}
